// GATEncoder_47725676593311
// MI455X (gfx1250) — matrix operand order repaired and checked by reading, not compiled, not run on hardware
//
#include <hip/hip_runtime.h>
#include <hip/hip_bf16.h>

typedef __attribute__((ext_vector_type(16))) _Float16 v16h;
typedef __attribute__((ext_vector_type(8)))  float    v8f;

#define IN_DIM 6
#define H1DIM 128   // HEADS1*HID = 4*32
#define H2DIM 64    // OUT_DIM

// ---------------------------------------------------------------------------
// Small prep kernels (O(N*d^2), negligible vs the N^2 attention pass)
// ---------------------------------------------------------------------------

__global__ void prep_feat1(const float* __restrict__ x, const float* __restrict__ W1,
                           float* __restrict__ hf32, _Float16* __restrict__ hf16, int n) {
  int id = blockIdx.x * blockDim.x + threadIdx.x;
  if (id >= n * H1DIM) return;
  int node = id >> 7, col = id & (H1DIM - 1);
  float s = 0.f;
#pragma unroll
  for (int k = 0; k < IN_DIM; ++k) s += x[node * IN_DIM + k] * W1[k * H1DIM + col];
  hf32[id] = s;
  hf16[id] = (_Float16)s;
}

__global__ void prep_e1(const float* __restrict__ hf32, const float* __restrict__ asrc,
                        const float* __restrict__ adst, float* __restrict__ esrc,
                        float* __restrict__ edst, int n) {
  int id = blockIdx.x * blockDim.x + threadIdx.x;
  if (id >= n * 4) return;
  int node = id >> 2, h = id & 3;
  float s = 0.f, d = 0.f;
#pragma unroll 8
  for (int k = 0; k < 32; ++k) {
    float v = hf32[node * H1DIM + h * 32 + k];
    s += v * asrc[h * 32 + k];
    d += v * adst[h * 32 + k];
  }
  esrc[id] = s;
  edst[id] = d;
}

__global__ void prep_feat2(const float* __restrict__ h1, const float* __restrict__ W2,
                           float* __restrict__ hf32, _Float16* __restrict__ hf16, int n) {
  int id = blockIdx.x * blockDim.x + threadIdx.x;
  if (id >= n * H2DIM) return;
  int node = id >> 6, col = id & (H2DIM - 1);
  float s = 0.f;
#pragma unroll 8
  for (int k = 0; k < H1DIM; ++k) s += h1[node * H1DIM + k] * W2[k * H2DIM + col];
  hf32[id] = s;
  hf16[id] = (_Float16)s;
}

__global__ void prep_e2(const float* __restrict__ hf32, const float* __restrict__ asrc,
                        const float* __restrict__ adst, float* __restrict__ esrc,
                        float* __restrict__ edst, int n) {
  int node = blockIdx.x * blockDim.x + threadIdx.x;
  if (node >= n) return;
  float s = 0.f, d = 0.f;
#pragma unroll 8
  for (int k = 0; k < H2DIM; ++k) {
    float v = hf32[node * H2DIM + k];
    s += v * asrc[k];
    d += v * adst[k];
  }
  esrc[node] = s;
  edst[node] = d;
}

// ---------------------------------------------------------------------------
// Fused masked-softmax attention + aggregation via V_WMMA_F32_16X16X32_F16.
// Softmax normalization is linear: accumulate unnormalized weighted sum and
// row-sum of weights in a single adjacency pass, divide at the end.
// ---------------------------------------------------------------------------

__device__ __forceinline__ float att_weight(float adjv, float es, float ed) {
  float z = es + ed;
  z = fmaxf(z, 0.2f * z);              // leaky_relu, slope 0.2
  float w = __expf(z);                 // |z| <~ 1 here: no max-subtraction needed
  return adjv > 0.f ? w : 0.f;
}

template <int HEADS, int D>
__global__ void __launch_bounds__(128)
gat_attn(const float* __restrict__ adj,      // [n, n]
         const _Float16* __restrict__ featB, // [n, HEADS*D] f16 (WMMA B staging)
         const float* __restrict__ e_src,    // [n*HEADS]
         const float* __restrict__ e_dst,    // [n*HEADS]
         const float* __restrict__ bias,     // [HEADS*D]
         float* __restrict__ out,            // [n, HEADS*D]
         int n, int apply_elu) {
  constexpr int FEAT = HEADS * D;
  constexpr int NTILES = FEAT / 16;   // 16-wide output column tiles
  constexpr int WAVES = 4;
  constexpr int TPW = NTILES / WAVES; // layer1: 2, layer2: 1

  // ping-pong LDS tiles: one barrier per j-chunk
  __shared__ float adjT[2][16][33];       // padded: conflict-free column reads
  __shared__ float edstT[2][HEADS][32];
  __shared__ float denomT[WAVES][16];

  const int tid  = threadIdx.x;
  const int wave = tid >> 5;
  const int lane = tid & 31;
  const int i0   = blockIdx.x * 16;
  const int row  = lane & 15;   // A-fragment M row / D-fragment N col
  const int hi   = lane >> 4;   // which K-half this lane holds (A layout)

  const int tt0  = wave * TPW;           // first column tile for this wave
  const int head = (tt0 * 16) / D;       // head this wave computes

  const float es = e_src[(i0 + row) * HEADS + head];

  // staging-thread coordinates (block-wide cooperative adj load)
  const int sf = tid * 4;                // 0..508
  const int sr = sf >> 5, sc = sf & 31;  // tile row / col of this thread's 4 floats
  // e_dst staging: HEADS=4 -> each wave stages one head; HEADS=1 -> all four
  // waves redundantly store the same value (benign same-value WAW, no branch).
  const int eh = (tid >> 5) % HEADS, ej = tid & 31;

  v8f acc[TPW];
#pragma unroll
  for (int t = 0; t < TPW; ++t)
#pragma unroll
    for (int v = 0; v < 8; ++v) acc[t][v] = 0.f;

  float dpart = 0.f;  // partial softmax denominator (row-sum of A halves)

  // ---- prologue: preload adj/e_dst tile 0 and B fragments for chunk 0 ----
  float4 areg = *(const float4*)(adj + (size_t)(i0 + sr) * n + sc);
  float  ereg = e_dst[ej * HEADS + eh];

  v16h Bb0[TPW], Bb1[TPW];
#pragma unroll
  for (int t = 0; t < TPW; ++t)
#pragma unroll
    for (int s = 0; s < 16; ++s)
      Bb0[t][s] = featB[(size_t)(hi * 8 + (s & 7) + (s >> 3) * 16) * FEAT + (tt0 + t) * 16 + row];

  // Process one 32-wide j-chunk; Bc holds this chunk's B fragments, Bn
  // receives the next chunk's (issued before the WMMAs so they overlap).
  auto do_chunk = [&](int jb, int p, v16h (&Bc)[TPW], v16h (&Bn)[TPW]) {
    const bool more = (jb + 32) < n;

    // ---- stage current tile (registers -> LDS), folding the self-loop
    //      a = max(adj, I) in branchlessly at store time ----
    {
      int di = (i0 + sr) - jb - sc;      // which of our 4 cols hits the diagonal
      float a0 = (di == 0) ? 1.f : areg.x;
      float a1 = (di == 1) ? 1.f : areg.y;
      float a2 = (di == 2) ? 1.f : areg.z;
      float a3 = (di == 3) ? 1.f : areg.w;
      adjT[p][sr][sc + 0] = a0; adjT[p][sr][sc + 1] = a1;
      adjT[p][sr][sc + 2] = a2; adjT[p][sr][sc + 3] = a3;
      edstT[p][eh][ej] = ereg;
    }

    // ---- kick off next tile's global loads (overlap with A-build + WMMA) ----
    if (more) {
      areg = *(const float4*)(adj + (size_t)(i0 + sr) * n + (jb + 32) + sc);
      ereg = e_dst[(jb + 32 + ej) * HEADS + eh];
      if (jb + 64 < n)   // distance-2 prefetch (global_prefetch_b8)
        __builtin_prefetch(adj + (size_t)(i0 + sr) * n + (jb + 64) + sc, 0, 1);
    }
    __syncthreads();

    // ---- build A fragment: 16x32 f16 attention weights for this head ----
    // A layout (16-bit, 16x32): lane L holds row M=L&15;
    //   lanes 0-15:  halves 0..7 -> K=0..7,  halves 8..15 -> K=16..23
    //   lanes 16-31: halves 0..7 -> K=8..15, halves 8..15 -> K=24..31
    v16h A;
    const int kb = hi * 8;
#pragma unroll
    for (int u = 0; u < 8; ++u) {
      int K0 = kb + u;
      int K1 = kb + 16 + u;
      float w0 = att_weight(adjT[p][row][K0], es, edstT[p][head][K0]);
      float w1 = att_weight(adjT[p][row][K1], es, edstT[p][head][K1]);
      A[u]     = (_Float16)w0;
      A[u + 8] = (_Float16)w1;
      dpart += w0 + w1;
    }

    // ---- issue next chunk's B fragment loads, then WMMA on current ----
    // B layout (16-bit, 32x16): lane L holds col N=L&15; halves in same K order as A.
    // featB is row-major [j, FEAT]: strided gather per lane.
    if (more) {
      const _Float16* bp = featB + (size_t)(jb + 32) * FEAT;
#pragma unroll
      for (int t = 0; t < TPW; ++t)
#pragma unroll
        for (int s = 0; s < 16; ++s)
          Bn[t][s] = bp[(size_t)(hi * 8 + (s & 7) + (s >> 3) * 16) * FEAT + (tt0 + t) * 16 + row];
    }
#pragma unroll
    for (int t = 0; t < TPW; ++t) {
      acc[t] = __builtin_amdgcn_wmma_f32_16x16x32_f16(
          /*neg_a=*/false, A, /*neg_b=*/false, Bc[t],
          /*c_mod=*/(short)0, acc[t], /*reuse_a=*/false, /*reuse_b=*/false);
    }
  };

  // unroll by 2: buffers alternate by name, no register-copy rotation
  for (int jb = 0; jb < n; jb += 64) {
    do_chunk(jb,      0, Bb0, Bb1);
    do_chunk(jb + 32, 1, Bb1, Bb0);
  }

  // ---- softmax denominator: combine the two K-halves of each row ----
  dpart += __shfl_xor(dpart, 16, 32);
  if (lane < 16) denomT[wave][lane] = dpart;
  __syncthreads();

  // ---- epilogue: normalize, bias, activation, store ----
  // C/D layout (32-bit, 16x16): VGPR v, lane L -> M = v + 8*(L>=16), N = L&15.
#pragma unroll
  for (int t = 0; t < TPW; ++t) {
    int tt = tt0 + t;
#pragma unroll
    for (int v = 0; v < 8; ++v) {
      int M = v + hi * 8;
      int c = tt * 16 + row;
      float val = acc[t][v] / denomT[wave][M] + bias[c];
      if (apply_elu) val = val > 0.f ? val : (__expf(val) - 1.f);
      out[(size_t)(i0 + M) * FEAT + c] = val;
    }
  }
}

// ---------------------------------------------------------------------------
// Launch
// ---------------------------------------------------------------------------
extern "C" void kernel_launch(void* const* d_in, const int* in_sizes, int n_in,
                              void* d_out, int out_size, void* d_ws, size_t ws_size,
                              hipStream_t stream) {
  const float* x     = (const float*)d_in[0];
  const float* adj   = (const float*)d_in[1];
  const float* W1    = (const float*)d_in[2];
  const float* asrc1 = (const float*)d_in[3];
  const float* adst1 = (const float*)d_in[4];
  const float* b1    = (const float*)d_in[5];
  const float* W2    = (const float*)d_in[6];
  const float* asrc2 = (const float*)d_in[7];
  const float* adst2 = (const float*)d_in[8];
  const float* b2    = (const float*)d_in[9];

  const int N = in_sizes[0] / IN_DIM;

  // workspace carve-up (256B aligned slices)
  char* ws = (char*)d_ws;
  size_t off = 0;
  auto take = [&](size_t bytes) -> char* {
    char* p = ws + off;
    off += (bytes + 255) & ~(size_t)255;
    return p;
  };
  float*    hf1_32 = (float*)take((size_t)N * H1DIM * 4);
  _Float16* hf1_16 = (_Float16*)take((size_t)N * H1DIM * 2);
  float*    es1    = (float*)take((size_t)N * 4 * 4);
  float*    ed1    = (float*)take((size_t)N * 4 * 4);
  float*    h1     = (float*)take((size_t)N * H1DIM * 4);
  float*    hf2_32 = (float*)take((size_t)N * H2DIM * 4);
  _Float16* hf2_16 = (_Float16*)take((size_t)N * H2DIM * 2);
  float*    es2    = (float*)take((size_t)N * 4);
  float*    ed2    = (float*)take((size_t)N * 4);

  // ---- layer 1 ----
  prep_feat1<<<(N * H1DIM + 255) / 256, 256, 0, stream>>>(x, W1, hf1_32, hf1_16, N);
  prep_e1<<<(N * 4 + 255) / 256, 256, 0, stream>>>(hf1_32, asrc1, adst1, es1, ed1, N);
  gat_attn<4, 32><<<N / 16, 128, 0, stream>>>(adj, hf1_16, es1, ed1, b1, h1, N,
                                              /*apply_elu=*/1);
  // ---- layer 2 ----
  prep_feat2<<<(N * H2DIM + 255) / 256, 256, 0, stream>>>(h1, W2, hf2_32, hf2_16, N);
  prep_e2<<<(N + 255) / 256, 256, 0, stream>>>(hf2_32, asrc2, adst2, es2, ed2, N);
  gat_attn<1, 64><<<N / 16, 128, 0, stream>>>(adj, hf2_16, es2, ed2, b2,
                                              (float*)d_out, N, /*apply_elu=*/0);
}
